// ClusterOrderedMamba_50826642981203
// MI455X (gfx1250) — compile-verified, matrix operand order checked by reading, not run on hardware
//
#include <hip/hip_runtime.h>
#include <hip/hip_bf16.h>
#include <math.h>

#define DM    512
#define DSZ   16
#define DC    4
#define DI    1024
#define DI4   256
#define KC    5
#define SEQ   1024
#define BATCH 4
#define NTOK  (BATCH * SEQ)   // 4096
#define XDIM  (2 * DSZ + 1)   // 33
#define TEMP_INV 2.0f

typedef __attribute__((ext_vector_type(16))) _Float16 v16h;
typedef __attribute__((ext_vector_type(8)))  _Float16 v8h;
typedef __attribute__((ext_vector_type(8)))  float    v8f;

__device__ __forceinline__ float sigmoidf_(float x) { return 1.f / (1.f + expf(-x)); }

// ---------------------------------------------------------------------------
// LDS-staged WMMA GEMM: C[4096,NG] = epilogue(A[4096,KG] @ W[KG,NG])
// EP: 0 = none, 1 = sigmoid(x + bias), 2 = x + resid, 3 = x + bias
// Block = 256 threads (8 waves), block tile = 64x64, wave tile = 16x32
// (one A fragment reused across two WMMAs). Tiles staged fp32->f16 in LDS;
// fragments read as two contiguous ds_load_b128 per matrix per lane.
// ---------------------------------------------------------------------------
template <int EP, int KG, int NG>
__global__ __launch_bounds__(256) void gemm_wmma_kernel(
    const float* __restrict__ A, const float* __restrict__ W,
    const float* __restrict__ bias, const float* __restrict__ resid,
    float* __restrict__ C)
{
  __shared__ __align__(16) _Float16 As[64 * 32];   // [row][k]
  __shared__ __align__(16) _Float16 Bt[64 * 32];   // [col][k] (transposed)

  const int tid  = threadIdx.x;
  const int lane = tid & 31;
  const int wave = tid >> 5;
  const int wx   = wave & 1;                 // 2 wave-cols (32 cols each)
  const int wy   = wave >> 1;                // 4 wave-rows (16 rows each)
  const int rowB = blockIdx.y * 64;
  const int colB = blockIdx.x * 64;
  const int mn   = lane & 15;
  const int hi   = lane >> 4;

  // staging assignments
  const int arow = tid >> 2;                 // 0..63
  const int acg  = (tid & 3) * 8;            // 0,8,16,24
  const int bkr  = tid >> 3;                 // 0..31
  const int bcg  = (tid & 7) * 8;            // 0..56

  v8f acc0 = {}, acc1 = {};

#pragma unroll 2
  for (int kk = 0; kk < KG; kk += 32) {
    // ---- stage A tile 64x32 (row-major f16) ----
    {
      const float* src = A + (size_t)(rowB + arow) * KG + kk + acg;
      __builtin_prefetch(src + 64, 0, 0);    // global_prefetch_b8
      const float4 f0 = *(const float4*)(src);
      const float4 f1 = *(const float4*)(src + 4);
      v8h h;
      h[0] = (_Float16)f0.x; h[1] = (_Float16)f0.y;
      h[2] = (_Float16)f0.z; h[3] = (_Float16)f0.w;
      h[4] = (_Float16)f1.x; h[5] = (_Float16)f1.y;
      h[6] = (_Float16)f1.z; h[7] = (_Float16)f1.w;
      *(v8h*)&As[arow * 32 + acg] = h;       // ds_store_b128
    }
    // ---- stage B tile 32x64, transposed into Bt[col][k] ----
    {
      const float* src = W + (size_t)(kk + bkr) * NG + colB + bcg;
      const float4 f0 = *(const float4*)(src);
      const float4 f1 = *(const float4*)(src + 4);
      const float fv[8] = {f0.x, f0.y, f0.z, f0.w, f1.x, f1.y, f1.z, f1.w};
#pragma unroll
      for (int e = 0; e < 8; ++e) Bt[(bcg + e) * 32 + bkr] = (_Float16)fv[e];
    }
    __syncthreads();
    // ---- read fragments: two b128 chunks per matrix per lane ----
    v16h af, bf0, bf1;
    {
      const v8h lo = *(const v8h*)&As[(wy * 16 + mn) * 32 + hi * 8];
      const v8h up = *(const v8h*)&As[(wy * 16 + mn) * 32 + 16 + hi * 8];
#pragma unroll
      for (int i = 0; i < 8; ++i) { af[i] = lo[i]; af[8 + i] = up[i]; }
    }
    {
      const int c0 = wx * 32 + mn;
      const v8h lo0 = *(const v8h*)&Bt[c0 * 32 + hi * 8];
      const v8h up0 = *(const v8h*)&Bt[c0 * 32 + 16 + hi * 8];
      const v8h lo1 = *(const v8h*)&Bt[(c0 + 16) * 32 + hi * 8];
      const v8h up1 = *(const v8h*)&Bt[(c0 + 16) * 32 + 16 + hi * 8];
#pragma unroll
      for (int i = 0; i < 8; ++i) {
        bf0[i] = lo0[i]; bf0[8 + i] = up0[i];
        bf1[i] = lo1[i]; bf1[8 + i] = up1[i];
      }
    }
    acc0 = __builtin_amdgcn_wmma_f32_16x16x32_f16(false, af, false, bf0,
                                                  (short)0, acc0, false, false);
    acc1 = __builtin_amdgcn_wmma_f32_16x16x32_f16(false, af, false, bf1,
                                                  (short)0, acc1, false, false);
    __syncthreads();
  }
  // ---- epilogue ----
  const int r0 = rowB + wy * 16;
  const int c0 = colB + wx * 32;
#pragma unroll
  for (int j = 0; j < 8; ++j) {
    const int r  = r0 + 8 * hi + j;
    const int ca = c0 + mn, cb = c0 + 16 + mn;
    float v0 = acc0[j], v1 = acc1[j];
    if (EP == 1)      { v0 = sigmoidf_(v0 + bias[ca]); v1 = sigmoidf_(v1 + bias[cb]); }
    else if (EP == 2) { v0 += resid[(size_t)r * NG + ca]; v1 += resid[(size_t)r * NG + cb]; }
    else if (EP == 3) { v0 += bias[ca]; v1 += bias[cb]; }
    C[(size_t)r * NG + ca] = v0;
    C[(size_t)r * NG + cb] = v1;
  }
}

// ---------------------------------------------------------------------------
// k-means
// ---------------------------------------------------------------------------
__global__ void bcast_centers_kernel(const float* __restrict__ cinit,
                                     float* __restrict__ centers)
{
  const int i = blockIdx.x * blockDim.x + threadIdx.x;
  if (i >= BATCH * KC * DM) return;
  centers[i] = cinit[i % (KC * DM)];
}

__global__ void kmeans_assign_kernel(const float* __restrict__ x,
                                     const float* __restrict__ centers,
                                     float* __restrict__ w,
                                     float* __restrict__ keys,
                                     float* __restrict__ mind, int finalpass)
{
  const int idx = blockIdx.x * blockDim.x + threadIdx.x;  // NTOK
  if (idx >= NTOK) return;
  const int b = idx / SEQ;
  const float* xp = x + (size_t)idx * DM;
  const float* cb = centers + (size_t)b * KC * DM;
  float d2[KC];
#pragma unroll
  for (int k = 0; k < KC; ++k) d2[k] = 0.f;
  for (int c = 0; c < DM; ++c) {
    const float xv = xp[c];
#pragma unroll
    for (int k = 0; k < KC; ++k) { const float df = xv - cb[k * DM + c]; d2[k] += df * df; }
  }
  float dist[KC], sc[KC], mx = -1e30f;
#pragma unroll
  for (int k = 0; k < KC; ++k) {
    dist[k] = sqrtf(d2[k] + 1e-12f);
    sc[k]   = -dist[k] * TEMP_INV;
    mx      = fmaxf(mx, sc[k]);
  }
  float sum = 0.f;
#pragma unroll
  for (int k = 0; k < KC; ++k) { sc[k] = expf(sc[k] - mx); sum += sc[k]; }
  float mw = -1.f, md = 1e30f; int ca = 0;
#pragma unroll
  for (int k = 0; k < KC; ++k) {
    const float wv = sc[k] / sum;
    w[(size_t)idx * KC + k] = wv;
    if (wv > mw) { mw = wv; ca = k; }
    md = fminf(md, dist[k]);
  }
  if (finalpass) {
    mind[idx] = md;
    keys[idx] = (float)ca * 1000.f + (1.f - mw);
  }
}

__global__ void kmeans_update_kernel(const float* __restrict__ x,
                                     const float* __restrict__ w,
                                     float* __restrict__ centers)
{
  const int idx = blockIdx.x * blockDim.x + threadIdx.x;   // BATCH*KC*DM
  if (idx >= BATCH * KC * DM) return;
  const int b = idx / (KC * DM);
  const int k = (idx / DM) % KC;
  const int c = idx % DM;
  float num = 0.f, den = 0.f;
  for (int n = 0; n < SEQ; ++n) {
    const size_t p = (size_t)b * SEQ + n;
    const float wv = w[p * KC + k];
    num += wv * x[p * DM + c];
    den += wv;
  }
  centers[idx] = num / (den + 1e-8f);
}

__global__ __launch_bounds__(1024) void importance_kernel(
    const float* __restrict__ mind, float* __restrict__ imp)
{
  __shared__ float red[SEQ];
  const int b = blockIdx.x, t = threadIdx.x;
  const float v = -mind[b * SEQ + t] * TEMP_INV;
  red[t] = v; __syncthreads();
  for (int s = SEQ / 2; s > 0; s >>= 1) {
    if (t < s) red[t] = fmaxf(red[t], red[t + s]);
    __syncthreads();
  }
  const float m = red[0]; __syncthreads();
  const float e = expf(v - m);
  red[t] = e; __syncthreads();
  for (int s = SEQ / 2; s > 0; s >>= 1) {
    if (t < s) red[t] += red[t + s];
    __syncthreads();
  }
  imp[b * SEQ + t] = e / red[0];
}

// ---------------------------------------------------------------------------
// per-batch bitonic argsort of 1024 keys (ascending)
// ---------------------------------------------------------------------------
__global__ __launch_bounds__(1024) void sort_kernel(
    const float* __restrict__ keys, int* __restrict__ ridx, int* __restrict__ inv)
{
  __shared__ float sk[SEQ];
  __shared__ int   si[SEQ];
  const int b = blockIdx.x, t = threadIdx.x;
  sk[t] = keys[b * SEQ + t];
  si[t] = t;
  __syncthreads();
  for (int ksz = 2; ksz <= SEQ; ksz <<= 1) {
    for (int j = ksz >> 1; j > 0; j >>= 1) {
      const int p = t ^ j;
      if (p > t) {
        const bool  up = ((t & ksz) == 0);
        const float ka = sk[t], kb = sk[p];
        if (up ? (ka > kb) : (ka < kb)) {
          sk[t] = kb; sk[p] = ka;
          const int ia = si[t]; si[t] = si[p]; si[p] = ia;
        }
      }
      __syncthreads();
    }
  }
  ridx[b * SEQ + t] = si[t];
  inv[b * SEQ + si[t]] = t;
}

// ---------------------------------------------------------------------------
// gather / flip helpers
// ---------------------------------------------------------------------------
__global__ void gather_rows_kernel(const float* __restrict__ in,
                                   const int* __restrict__ idx,
                                   float* __restrict__ out)
{
  const int i = blockIdx.x * blockDim.x + threadIdx.x;   // NTOK*DM
  if (i >= NTOK * DM) return;
  const int p = i / DM, c = i % DM;
  const int b = p / SEQ;
  out[i] = in[((size_t)b * SEQ + idx[p]) * DM + c];
}

__global__ void gather_scalar_kernel(const float* __restrict__ in,
                                     const int* __restrict__ idx,
                                     float* __restrict__ out)
{
  const int p = blockIdx.x * blockDim.x + threadIdx.x;
  if (p >= NTOK) return;
  const int b = p / SEQ;
  out[p] = in[b * SEQ + idx[p]];
}

__global__ void flip_rows_kernel(const float* __restrict__ in, float* __restrict__ out)
{
  const int i = blockIdx.x * blockDim.x + threadIdx.x;   // NTOK*DM
  if (i >= NTOK * DM) return;
  const int p = i / DM, c = i % DM;
  const int b = p / SEQ, n = p % SEQ;
  out[i] = in[((size_t)b * SEQ + (SEQ - 1 - n)) * DM + c];
}

__global__ void flip_scalar_kernel(const float* __restrict__ in, float* __restrict__ out)
{
  const int p = blockIdx.x * blockDim.x + threadIdx.x;
  if (p >= NTOK) return;
  const int b = p / SEQ, n = p % SEQ;
  out[p] = in[b * SEQ + (SEQ - 1 - n)];
}

__global__ void concat_kernel(const float* __restrict__ xf, const float* __restrict__ xb,
                              float* __restrict__ cat)
{
  const int i = blockIdx.x * blockDim.x + threadIdx.x;   // NTOK*2*DM
  if (i >= NTOK * 2 * DM) return;
  const int p = i / (2 * DM), c = i % (2 * DM);
  cat[i] = (c < DM) ? xf[(size_t)p * DM + c] : xb[(size_t)p * DM + (c - DM)];
}

// ---------------------------------------------------------------------------
// layernorm (one 256-thread block per token, DM=512 -> 2 elems/thread)
// ---------------------------------------------------------------------------
__global__ __launch_bounds__(256) void layernorm_kernel(
    const float* __restrict__ x, const float* __restrict__ g,
    const float* __restrict__ bb, float* __restrict__ out)
{
  __shared__ float s1[256], s2[256];
  const int p = blockIdx.x, t = threadIdx.x;
  const float a  = x[(size_t)p * DM + t];
  const float b2 = x[(size_t)p * DM + 256 + t];
  s1[t] = a + b2;
  s2[t] = a * a + b2 * b2;
  __syncthreads();
  for (int s = 128; s > 0; s >>= 1) {
    if (t < s) { s1[t] += s1[t + s]; s2[t] += s2[t + s]; }
    __syncthreads();
  }
  const float mu  = s1[0] * (1.f / DM);
  const float var = s2[0] * (1.f / DM) - mu * mu;
  const float inv = rsqrtf(var + 1e-5f);
  out[(size_t)p * DM + t]       = (a  - mu) * inv * g[t]       + bb[t];
  out[(size_t)p * DM + 256 + t] = (b2 - mu) * inv * g[256 + t] + bb[256 + t];
}

// ---------------------------------------------------------------------------
// depthwise causal conv (DC=4) + silu
// ---------------------------------------------------------------------------
__global__ void conv_silu_kernel(const float* __restrict__ xz,
                                 const float* __restrict__ cw,
                                 const float* __restrict__ cb,
                                 float* __restrict__ xc)
{
  const int i = blockIdx.x * blockDim.x + threadIdx.x;   // NTOK*DI
  if (i >= NTOK * DI) return;
  const int d = i % DI, p = i / DI;
  const int b = p / SEQ, t = p % SEQ;
  float acc = cb[d];
#pragma unroll
  for (int j = 0; j < DC; ++j) {
    const int tt = t + j - (DC - 1);
    if (tt >= 0) acc += xz[((size_t)b * SEQ + tt) * (2 * DI) + d] * cw[d * DC + j];
  }
  xc[i] = acc * sigmoidf_(acc);
}

__global__ void xssm_kernel(const float* __restrict__ xc,
                            const float* __restrict__ xpw,
                            float* __restrict__ xssm)
{
  const int i = blockIdx.x * blockDim.x + threadIdx.x;   // NTOK*33
  if (i >= NTOK * XDIM) return;
  const int p = i / XDIM, o = i % XDIM;
  float acc = 0.f;
  for (int d = 0; d < DI; ++d) acc += xc[(size_t)p * DI + d] * xpw[d * XDIM + o];
  xssm[i] = acc;
}

__global__ void h1_kernel(const float* __restrict__ imp, const float* __restrict__ w1,
                          const float* __restrict__ b1, float* __restrict__ out)
{
  const int i = blockIdx.x * blockDim.x + threadIdx.x;   // NTOK*DI4
  if (i >= NTOK * DI4) return;
  const int p = i / DI4, j = i % DI4;
  const float v = imp[p] * w1[j] + b1[j];
  out[i] = v > 0.f ? v : 0.f;
}

// delta = softplus(draw*dt_w + dt_b) * (1 + mod); in-place on mod
__global__ void delta_kernel(const float* __restrict__ xssm,
                             const float* __restrict__ dtw,
                             const float* __restrict__ dtb,
                             float* __restrict__ mod)
{
  const int i = blockIdx.x * blockDim.x + threadIdx.x;   // NTOK*DI
  if (i >= NTOK * DI) return;
  const int p = i / DI, d = i % DI;
  const float u  = xssm[(size_t)p * XDIM] * dtw[d] + dtb[d];
  const float sp = (u > 20.f) ? u : log1pf(expf(u));
  mod[i] = sp * (1.f + mod[i]);
}

// ---------------------------------------------------------------------------
// selective-scan: each thread owns one channel d, h[16] in registers,
// B_t / C_t staged through LDS every step.  Also fuses +xc*D_p and *silu(z).
// ---------------------------------------------------------------------------
__global__ __launch_bounds__(256) void scan_kernel(
    const float* __restrict__ delta, const float* __restrict__ xssm,
    const float* __restrict__ xc, const float* __restrict__ xz,
    const float* __restrict__ Alog, const float* __restrict__ Dp,
    float* __restrict__ y)
{
  const int d = blockIdx.x * 256 + threadIdx.x;
  const int b = blockIdx.y;
  __shared__ float Bs[DSZ], Cs[DSZ];
  float a[DSZ], h[DSZ];
#pragma unroll
  for (int s = 0; s < DSZ; ++s) { a[s] = -expf(Alog[d * DSZ + s]); h[s] = 0.f; }
  const float dp = Dp[d];
  for (int t = 0; t < SEQ; ++t) {
    const int p = b * SEQ + t;
    if (threadIdx.x < DSZ)          Bs[threadIdx.x]       = xssm[p * XDIM + 1 + threadIdx.x];
    else if (threadIdx.x < 2 * DSZ) Cs[threadIdx.x - DSZ] = xssm[p * XDIM + 1 + threadIdx.x];
    __syncthreads();
    const float dt = delta[(size_t)p * DI + d];
    const float xt = xc[(size_t)p * DI + d];
    float yv = 0.f;
#pragma unroll
    for (int s = 0; s < DSZ; ++s) {
      h[s] = expf(dt * a[s]) * h[s] + dt * Bs[s] * xt;
      yv  += h[s] * Cs[s];
    }
    const float z = xz[(size_t)p * (2 * DI) + DI + d];
    y[(size_t)p * DI + d] = (yv + xt * dp) * (z * sigmoidf_(z));
    __syncthreads();
  }
}

// ---------------------------------------------------------------------------
// orchestration
// ---------------------------------------------------------------------------
struct MambaParams {
  const float *in_w, *conv_w, *conv_b, *xp_w, *dt_w, *dt_b,
              *cm_w1, *cm_b1, *cm_w2, *cm_b2, *A_log, *D_p, *out_w, *ln_g, *ln_b;
};
struct MambaScratch { float *xn, *xz, *xc, *modb, *ybuf, *h1b, *xssmb; };

static void run_mamba_layer(const float* xin, float* xout, const float* imp, int li,
                            const MambaParams& P, const MambaScratch& S, hipStream_t st)
{
  const int TB = 256;
  layernorm_kernel<<<NTOK, TB, 0, st>>>(xin, P.ln_g + li * DM, P.ln_b + li * DM, S.xn);
  gemm_wmma_kernel<0, DM, 2 * DI><<<dim3((2 * DI) / 64, NTOK / 64), TB, 0, st>>>(
      S.xn, P.in_w + (size_t)li * DM * 2 * DI, nullptr, nullptr, S.xz);
  conv_silu_kernel<<<(NTOK * DI + TB - 1) / TB, TB, 0, st>>>(
      S.xz, P.conv_w + (size_t)li * DI * DC, P.conv_b + (size_t)li * DI, S.xc);
  xssm_kernel<<<(NTOK * XDIM + TB - 1) / TB, TB, 0, st>>>(
      S.xc, P.xp_w + (size_t)li * DI * XDIM, S.xssmb);
  h1_kernel<<<(NTOK * DI4 + TB - 1) / TB, TB, 0, st>>>(
      imp, P.cm_w1 + (size_t)li * DI4, P.cm_b1 + (size_t)li * DI4, S.h1b);
  gemm_wmma_kernel<1, DI4, DI><<<dim3(DI / 64, NTOK / 64), TB, 0, st>>>(
      S.h1b, P.cm_w2 + (size_t)li * DI4 * DI, P.cm_b2 + (size_t)li * DI, nullptr, S.modb);
  delta_kernel<<<(NTOK * DI + TB - 1) / TB, TB, 0, st>>>(
      S.xssmb, P.dt_w + (size_t)li * DI, P.dt_b + (size_t)li * DI, S.modb);
  scan_kernel<<<dim3(DI / 256, BATCH), 256, 0, st>>>(
      S.modb, S.xssmb, S.xc, S.xz, P.A_log + (size_t)li * DI * DSZ,
      P.D_p + (size_t)li * DI, S.ybuf);
  gemm_wmma_kernel<2, DI, DM><<<dim3(DM / 64, NTOK / 64), TB, 0, st>>>(
      S.ybuf, P.out_w + (size_t)li * DI * DM, nullptr, xin, xout);
}

extern "C" void kernel_launch(void* const* d_in, const int* in_sizes, int n_in,
                              void* d_out, int out_size, void* d_ws, size_t ws_size,
                              hipStream_t stream)
{
  (void)in_sizes; (void)n_in; (void)out_size; (void)ws_size;
  const float* x     = (const float*)d_in[0];
  const float* cinit = (const float*)d_in[1];
  MambaParams P;
  P.in_w   = (const float*)d_in[2];
  P.conv_w = (const float*)d_in[3];
  P.conv_b = (const float*)d_in[4];
  P.xp_w   = (const float*)d_in[5];
  P.dt_w   = (const float*)d_in[6];
  P.dt_b   = (const float*)d_in[7];
  P.cm_w1  = (const float*)d_in[8];
  P.cm_b1  = (const float*)d_in[9];
  P.cm_w2  = (const float*)d_in[10];
  P.cm_b2  = (const float*)d_in[11];
  P.A_log  = (const float*)d_in[12];
  P.D_p    = (const float*)d_in[13];
  P.out_w  = (const float*)d_in[14];
  P.ln_g   = (const float*)d_in[15];
  P.ln_b   = (const float*)d_in[16];
  const float* fusion_w = (const float*)d_in[17];
  const float* fusion_b = (const float*)d_in[18];
  const float* fn_g     = (const float*)d_in[19];
  const float* fn_b     = (const float*)d_in[20];

  float* ws = (float*)d_ws;
  size_t off = 0;
  auto alloc = [&](size_t n) { float* p = ws + off; off += n; return p; };
  float* centers = alloc((size_t)BATCH * KC * DM);
  float* wclu    = alloc((size_t)NTOK * KC);
  float* keysb   = alloc(NTOK);
  float* mind    = alloc(NTOK);
  float* imp     = alloc(NTOK);
  float* rimp    = alloc(NTOK);
  float* fimp    = alloc(NTOK);
  int*   ridx    = (int*)alloc(NTOK);
  int*   inv     = (int*)alloc(NTOK);
  float* rx      = alloc((size_t)NTOK * DM);
  float* s0      = alloc((size_t)NTOK * DM);
  float* s1      = alloc((size_t)NTOK * DM);
  float* s2      = alloc((size_t)NTOK * DM);
  MambaScratch S;
  S.xn    = alloc((size_t)NTOK * DM);
  S.xz    = alloc((size_t)NTOK * 2 * DI);
  S.xc    = alloc((size_t)NTOK * DI);
  S.modb  = alloc((size_t)NTOK * DI);
  S.ybuf  = alloc((size_t)NTOK * DI);
  S.h1b   = alloc((size_t)NTOK * DI4);
  S.xssmb = alloc((size_t)NTOK * XDIM);

  const int TB = 256;
  // ---- soft k-means (3 updates + final assignment) ----
  bcast_centers_kernel<<<(BATCH * KC * DM + TB - 1) / TB, TB, 0, stream>>>(cinit, centers);
  for (int it = 0; it < 3; ++it) {
    kmeans_assign_kernel<<<(NTOK + TB - 1) / TB, TB, 0, stream>>>(
        x, centers, wclu, keysb, mind, 0);
    kmeans_update_kernel<<<(BATCH * KC * DM + TB - 1) / TB, TB, 0, stream>>>(
        x, wclu, centers);
  }
  kmeans_assign_kernel<<<(NTOK + TB - 1) / TB, TB, 0, stream>>>(
      x, centers, wclu, keysb, mind, 1);
  importance_kernel<<<BATCH, SEQ, 0, stream>>>(mind, imp);
  // ---- sort + reorder ----
  sort_kernel<<<BATCH, SEQ, 0, stream>>>(keysb, ridx, inv);
  gather_rows_kernel<<<(NTOK * DM + TB - 1) / TB, TB, 0, stream>>>(x, ridx, rx);
  gather_scalar_kernel<<<(NTOK + TB - 1) / TB, TB, 0, stream>>>(imp, ridx, rimp);
  // ---- forward layers (result -> s0), rx preserved ----
  run_mamba_layer(rx, s1, rimp, 0, P, S, stream);
  run_mamba_layer(s1, s0, rimp, 1, P, S, stream);
  // ---- backward layers on flipped sequence ----
  flip_rows_kernel<<<(NTOK * DM + TB - 1) / TB, TB, 0, stream>>>(rx, s1);
  flip_scalar_kernel<<<(NTOK + TB - 1) / TB, TB, 0, stream>>>(rimp, fimp);
  run_mamba_layer(s1, s2, fimp, 2, P, S, stream);
  run_mamba_layer(s2, s1, fimp, 3, P, S, stream);
  flip_rows_kernel<<<(NTOK * DM + TB - 1) / TB, TB, 0, stream>>>(s1, s2);   // xb -> s2
  // ---- fusion + restore + final LN ----
  concat_kernel<<<(NTOK * 2 * DM + TB - 1) / TB, TB, 0, stream>>>(s0, s2, S.xz);
  gemm_wmma_kernel<3, 2 * DM, DM><<<dim3(DM / 64, NTOK / 64), TB, 0, stream>>>(
      S.xz, fusion_w, fusion_b, nullptr, S.xc);
  gather_rows_kernel<<<(NTOK * DM + TB - 1) / TB, TB, 0, stream>>>(S.xc, inv, S.xn);
  layernorm_kernel<<<NTOK, TB, 0, stream>>>(S.xn, fn_g, fn_b, (float*)d_out);
}
